// NeuralNetwork_SA_9216999817611
// MI455X (gfx1250) — matrix operand order repaired and checked by reading, not compiled, not run on hardware
//
#include <hip/hip_runtime.h>
#include <hip/hip_fp16.h>
#include <math.h>

// ---------------------------------------------------------------------------
// Problem dims (fixed by the reference)
// ---------------------------------------------------------------------------
#define BATCH   16384
#define IN_DIM  1284
#define KPAD    1312      // 1284 padded up to a multiple of 32 (41*32)
#define HID2    1024      // 2*HID
#define HID     512
#define HDIM    256

typedef __attribute__((ext_vector_type(16))) _Float16 v16h;
typedef __attribute__((ext_vector_type(8)))  float    v8f;

// ---------------------------------------------------------------------------
// WMMA GEMM:  C[M,N] = A[M,K](f16) @ W[N,K](f16)^T + bias[N](f32)
// Tile: 128x128 per block (256 threads = 8 wave32), wave grid 2(M) x 4(N),
// each wave owns a 64x32 region = 4x2 tiles of v_wmma_f32_16x16x32_f16.
// Double-buffered LDS: global loads for tile t+1 are issued before the WMMA
// block of tile t, one barrier per K-step, so HBM/L2 latency hides under
// matrix math.
// MODE 0: store f32.  MODE 1: store tanh(x) as f16 (gate hidden layer).
// ---------------------------------------------------------------------------
#define TILE_M 128
#define TILE_N 128
#define TILE_K 32
#define LDSS   40         // 32 + 8 f16 pad to break bank alignment

union FragU { uint4 q[2]; v16h h; };

template<int MODE>
__global__ __launch_bounds__(256) void gemm_f16_wmma(
    const _Float16* __restrict__ A, int lda,
    const _Float16* __restrict__ W, int ldw,
    const float* __restrict__ bias,
    void* __restrict__ Cout, int ldc, int K)
{
  __shared__ _Float16 As[2][TILE_M * LDSS];
  __shared__ _Float16 Ws[2][TILE_N * LDSS];

  const int tid  = threadIdx.x;
  const int lane = tid & 31;
  const int wid  = tid >> 5;
  const int wm   = wid & 1;          // wave M index (0..1)
  const int wn   = wid >> 1;         // wave N index (0..3)
  const int row0 = blockIdx.y * TILE_M;
  const int col0 = blockIdx.x * TILE_N;
  const int half = lane >> 4;        // 0 for lanes 0-15, 1 for lanes 16-31
  const int l16  = lane & 15;

  // Per-thread cooperative-load geometry: 2 chunks of 8 f16 (16B) per matrix.
  //   chunk c covers row ldr + 64*c, cols ldk..ldk+7 of the 128x32 tile.
  const int ldr = tid >> 2;          // 0..63
  const int ldk = (tid & 3) * 8;     // 0,8,16,24
  const _Float16* gA = A + (size_t)(row0 + ldr) * lda + ldk;
  const _Float16* gW = W + (size_t)(col0 + ldr) * ldw + ldk;
  const size_t sA64 = (size_t)64 * lda;
  const size_t sW64 = (size_t)64 * ldw;
  const int ls0 = ldr * LDSS + ldk;
  const int ls1 = (ldr + 64) * LDSS + ldk;

  v8f acc[4][2];
  const v8f vz = {0.f,0.f,0.f,0.f,0.f,0.f,0.f,0.f};
#pragma unroll
  for (int m = 0; m < 4; ++m)
#pragma unroll
    for (int n = 0; n < 2; ++n) acc[m][n] = vz;

  // ---- prologue: fill buffer 0 with tile 0 ----
  uint4 ra0 = *(const uint4*)(gA);
  uint4 ra1 = *(const uint4*)(gA + sA64);
  uint4 rw0 = *(const uint4*)(gW);
  uint4 rw1 = *(const uint4*)(gW + sW64);
  *(uint4*)(&As[0][ls0]) = ra0;
  *(uint4*)(&As[0][ls1]) = ra1;
  *(uint4*)(&Ws[0][ls0]) = rw0;
  *(uint4*)(&Ws[0][ls1]) = rw1;
  __syncthreads();

  const int nt = K / TILE_K;
  for (int t = 0; t < nt; ++t) {
    const int cur  = t & 1;
    const bool more = (t + 1 < nt);

    // Issue global loads for tile t+1 now; they retire during the WMMAs.
    if (more) {
      const _Float16* pA = gA + (size_t)(t + 1) * TILE_K;
      const _Float16* pW = gW + (size_t)(t + 1) * TILE_K;
      ra0 = *(const uint4*)(pA);
      ra1 = *(const uint4*)(pA + sA64);
      rw0 = *(const uint4*)(pW);
      rw1 = *(const uint4*)(pW + sW64);
      if (t + 2 < nt) {   // gfx1250 global_prefetch_b8, two tiles ahead
        __builtin_prefetch(pA + TILE_K, 0, 3);
        __builtin_prefetch(pW + TILE_K, 0, 3);
      }
    }

    // A fragment (16-bit A 16x32, ISA 7.12.2): lane half h, row M=l16,
    //   elems 0..7  -> K = 8h + {0..7},  elems 8..15 -> K = 16 + 8h + {0..7}
    v16h afr[4];
#pragma unroll
    for (int m = 0; m < 4; ++m) {
      const _Float16* p = &As[cur][(wm * 64 + m * 16 + l16) * LDSS];
      FragU u;
      u.q[0] = *(const uint4*)(p + half * 8);
      u.q[1] = *(const uint4*)(p + 16 + half * 8);
      afr[m] = u.h;
    }
    // B fragment (16-bit B 32x16): lane col N=l16,
    //   elems 0..7  -> K = 8h + {0..7},  elems 8..15 -> K = 16 + 8h + {0..7}
    v16h bfr[2];
#pragma unroll
    for (int n = 0; n < 2; ++n) {
      const _Float16* p = &Ws[cur][(wn * 32 + n * 16 + l16) * LDSS];
      FragU u;
      u.q[0] = *(const uint4*)(p + half * 8);
      u.q[1] = *(const uint4*)(p + 16 + half * 8);
      bfr[n] = u.h;
    }

#pragma unroll
    for (int m = 0; m < 4; ++m)
#pragma unroll
      for (int n = 0; n < 2; ++n)
        acc[m][n] = __builtin_amdgcn_wmma_f32_16x16x32_f16(
            false, afr[m], false, bfr[n], (short)0, acc[m][n], false, false);

    // Park tile t+1 into the other buffer (nobody reads it this iteration).
    if (more) {
      const int nb = cur ^ 1;
      *(uint4*)(&As[nb][ls0]) = ra0;
      *(uint4*)(&As[nb][ls1]) = ra1;
      *(uint4*)(&Ws[nb][ls0]) = rw0;
      *(uint4*)(&Ws[nb][ls1]) = rw1;
    }
    __syncthreads();
  }

  // Epilogue: C/D layout — VGPR r, lanes 0-15: (M=r, N=lane); lanes 16-31: M=r+8
#pragma unroll
  for (int m = 0; m < 4; ++m) {
#pragma unroll
    for (int n = 0; n < 2; ++n) {
      const int ccol = col0 + wn * 32 + n * 16 + l16;
      const float bv = bias[ccol];
#pragma unroll
      for (int r = 0; r < 8; ++r) {
        const int crow = row0 + wm * 64 + m * 16 + r + half * 8;
        const float v = acc[m][n][r] + bv;
        if (MODE == 0) {
          ((float*)Cout)[(size_t)crow * ldc + ccol] = v;
        } else {
          ((_Float16*)Cout)[(size_t)crow * ldc + ccol] = (_Float16)tanhf(v);
        }
      }
    }
  }
}

// ---------------------------------------------------------------------------
// ctx = concat(state, task, action) -> f16, K padded 1284 -> 1312 with zeros
// ---------------------------------------------------------------------------
__global__ __launch_bounds__(256) void pack_ctx(
    const float* __restrict__ state, const float* __restrict__ task,
    const float* __restrict__ action, _Float16* __restrict__ out)
{
  size_t i = (size_t)blockIdx.x * blockDim.x + threadIdx.x;
  const size_t total = (size_t)BATCH * KPAD;
  if (i >= total) return;
  const int b = (int)(i / KPAD);
  const int c = (int)(i % KPAD);
  float v;
  if (c < 1024)       v = state[(size_t)b * 1024 + c];
  else if (c < 1028)  v = task[(size_t)b * 4 + (c - 1024)];
  else if (c < 1284)  v = action[(size_t)b * 256 + (c - 1028)];
  else                v = 0.0f;
  out[i] = (_Float16)v;
}

// fp32 [N,K] weight -> f16 [N,KP] zero-padded
__global__ __launch_bounds__(256) void convert_w(
    const float* __restrict__ src, _Float16* __restrict__ dst, int N, int K, int KP)
{
  size_t i = (size_t)blockIdx.x * blockDim.x + threadIdx.x;
  const size_t total = (size_t)N * KP;
  if (i >= total) return;
  const int r = (int)(i / KP);
  const int c = (int)(i % KP);
  dst[i] = (c < K) ? (_Float16)src[(size_t)r * K + c] : (_Float16)0.0f;
}

// ---------------------------------------------------------------------------
// Per-row argmax (first-max-index tie-break, matches jnp.argmax).
// argmax(softmax(z)) == argmax(z) so the softmax is elided entirely.
// ---------------------------------------------------------------------------
__global__ __launch_bounds__(256) void rowargmax(
    const float* __restrict__ x, int N, unsigned* __restrict__ kout)
{
  const int row = blockIdx.x;
  const float* p = x + (size_t)row * N;
  float best = -3.402823466e38f;
  int   bi   = 0x7FFFFFFF;
  for (int i = threadIdx.x; i < N; i += 256) {
    const float v = p[i];
    if (v > best) { best = v; bi = i; }   // strided i is increasing per thread
  }
  __shared__ float sv[256];
  __shared__ int   si[256];
  sv[threadIdx.x] = best; si[threadIdx.x] = bi;
  __syncthreads();
  for (int s = 128; s > 0; s >>= 1) {
    if (threadIdx.x < (unsigned)s) {
      const float ov = sv[threadIdx.x + s];
      const int   oi = si[threadIdx.x + s];
      if (ov > sv[threadIdx.x] || (ov == sv[threadIdx.x] && oi < si[threadIdx.x])) {
        sv[threadIdx.x] = ov; si[threadIdx.x] = oi;
      }
    }
    __syncthreads();
  }
  if (threadIdx.x == 0) kout[row] = (unsigned)si[0];
}

// ---------------------------------------------------------------------------
// kwta: stable descending rank; keep rank<k else damp by 1/3; emit f16
// (next GEMM input). One block per row, N <= 1024.
// ---------------------------------------------------------------------------
__global__ __launch_bounds__(256) void kwta_kernel(
    const float* __restrict__ x, const unsigned* __restrict__ kk,
    _Float16* __restrict__ out, int N)
{
  const int row = blockIdx.x;
  __shared__ float rb[1024];
  const float* p = x + (size_t)row * N;
  for (int i = threadIdx.x; i < N; i += 256) rb[i] = p[i];
  __syncthreads();
  const unsigned kv = kk[row];
  for (int i = threadIdx.x; i < N; i += 256) {
    const float xi = rb[i];
    unsigned rank = 0;
    for (int j = 0; j < N; ++j) {
      const float xj = rb[j];
      rank += (xj > xi) || (xj == xi && j < i);   // stable argsort rank
    }
    const float r = (rank < kv) ? xi : xi / 3.0f;
    out[(size_t)row * N + i] = (_Float16)r;
  }
}

// ---------------------------------------------------------------------------
// Host-side orchestration
// ---------------------------------------------------------------------------
static inline dim3 gemm_grid(int M, int N) { return dim3(N / TILE_N, M / TILE_M); }

extern "C" void kernel_launch(void* const* d_in, const int* in_sizes, int n_in,
                              void* d_out, int out_size, void* d_ws, size_t ws_size,
                              hipStream_t stream) {
  (void)in_sizes; (void)n_in; (void)out_size; (void)ws_size;

  const float* state  = (const float*)d_in[0];
  const float* task   = (const float*)d_in[1];
  const float* action = (const float*)d_in[2];
  const float* W_c11 = (const float*)d_in[3];  const float* b_c11 = (const float*)d_in[4];
  const float* W_c12 = (const float*)d_in[5];  const float* b_c12 = (const float*)d_in[6];
  const float* W_c21 = (const float*)d_in[7];  const float* b_c21 = (const float*)d_in[8];
  const float* W_c22 = (const float*)d_in[9];  const float* b_c22 = (const float*)d_in[10];
  const float* W_c31 = (const float*)d_in[11]; const float* b_c31 = (const float*)d_in[12];
  const float* W_c32 = (const float*)d_in[13]; const float* b_c32 = (const float*)d_in[14];
  const float* W_l1  = (const float*)d_in[15]; const float* b_l1  = (const float*)d_in[16];
  const float* W_l2  = (const float*)d_in[17]; const float* b_l2  = (const float*)d_in[18];
  const float* W_l3  = (const float*)d_in[19]; const float* b_l3  = (const float*)d_in[20];
  const float* W_l4  = (const float*)d_in[21]; const float* b_l4  = (const float*)d_in[22];

  // ---- workspace layout (256B aligned, ~181 MB total) ----
  char* ws = (char*)d_ws;
  size_t off = 0;
  auto alloc = [&](size_t bytes) -> char* {
    char* p = ws + off;
    off = (off + bytes + 255) & ~(size_t)255;
    return p;
  };
  _Float16* ctx16 = (_Float16*)alloc((size_t)BATCH * KPAD * 2);
  _Float16* wc11  = (_Float16*)alloc((size_t)HID2 * KPAD * 2);
  _Float16* wc12  = (_Float16*)alloc((size_t)HID2 * HID2 * 2);
  _Float16* wc21  = (_Float16*)alloc((size_t)HID  * KPAD * 2);
  _Float16* wc22  = (_Float16*)alloc((size_t)HID  * HID  * 2);
  _Float16* wc31  = (_Float16*)alloc((size_t)HDIM * KPAD * 2);
  _Float16* wc32  = (_Float16*)alloc((size_t)HDIM * HDIM * 2);
  _Float16* wl1   = (_Float16*)alloc((size_t)HID2 * KPAD * 2);
  _Float16* wl2   = (_Float16*)alloc((size_t)HID  * HID2 * 2);
  _Float16* wl3   = (_Float16*)alloc((size_t)HDIM * HID  * 2);
  _Float16* wl4   = (_Float16*)alloc((size_t)HDIM * HDIM * 2);
  _Float16* h16   = (_Float16*)alloc((size_t)BATCH * HID2 * 2); // gate hidden, reused as x1_16
  float*    fbuf  = (float*)   alloc((size_t)BATCH * HID2 * 4); // logits / pre-kwta x
  _Float16* x2_16 = (_Float16*)alloc((size_t)BATCH * HID  * 2);
  _Float16* x3_16 = (_Float16*)alloc((size_t)BATCH * HDIM * 2);
  unsigned* k1    = (unsigned*)alloc((size_t)BATCH * 4);
  unsigned* k2    = (unsigned*)alloc((size_t)BATCH * 4);
  unsigned* k3    = (unsigned*)alloc((size_t)BATCH * 4);

  // ---- stage 0: pack ctx + convert weights to f16 (K zero-padded) ----
  {
    const size_t tot = (size_t)BATCH * KPAD;
    pack_ctx<<<dim3((unsigned)((tot + 255) / 256)), dim3(256), 0, stream>>>(
        state, task, action, ctx16);
  }
  auto conv = [&](const float* src, _Float16* dst, int N, int K, int KP) {
    const size_t tot = (size_t)N * KP;
    convert_w<<<dim3((unsigned)((tot + 255) / 256)), dim3(256), 0, stream>>>(
        src, dst, N, K, KP);
  };
  conv(W_c11, wc11, HID2, IN_DIM, KPAD);
  conv(W_c12, wc12, HID2, HID2,  HID2);
  conv(W_c21, wc21, HID,  IN_DIM, KPAD);
  conv(W_c22, wc22, HID,  HID,   HID);
  conv(W_c31, wc31, HDIM, IN_DIM, KPAD);
  conv(W_c32, wc32, HDIM, HDIM,  HDIM);
  conv(W_l1,  wl1,  HID2, IN_DIM, KPAD);
  conv(W_l2,  wl2,  HID,  HID2,  HID2);
  conv(W_l3,  wl3,  HDIM, HID,   HID);
  conv(W_l4,  wl4,  HDIM, HDIM,  HDIM);

  // ---- gates: tanh GEMM -> logits GEMM -> argmax (softmax elided) ----
  // gate 1 (N=1024)
  gemm_f16_wmma<1><<<gemm_grid(BATCH, HID2), 256, 0, stream>>>(
      ctx16, KPAD, wc11, KPAD, b_c11, h16, HID2, KPAD);
  gemm_f16_wmma<0><<<gemm_grid(BATCH, HID2), 256, 0, stream>>>(
      h16, HID2, wc12, HID2, b_c12, fbuf, HID2, HID2);
  rowargmax<<<dim3(BATCH), 256, 0, stream>>>(fbuf, HID2, k1);
  // gate 2 (N=512)
  gemm_f16_wmma<1><<<gemm_grid(BATCH, HID), 256, 0, stream>>>(
      ctx16, KPAD, wc21, KPAD, b_c21, h16, HID, KPAD);
  gemm_f16_wmma<0><<<gemm_grid(BATCH, HID), 256, 0, stream>>>(
      h16, HID, wc22, HID, b_c22, fbuf, HID, HID);
  rowargmax<<<dim3(BATCH), 256, 0, stream>>>(fbuf, HID, k2);
  // gate 3 (N=256)
  gemm_f16_wmma<1><<<gemm_grid(BATCH, HDIM), 256, 0, stream>>>(
      ctx16, KPAD, wc31, KPAD, b_c31, h16, HDIM, KPAD);
  gemm_f16_wmma<0><<<gemm_grid(BATCH, HDIM), 256, 0, stream>>>(
      h16, HDIM, wc32, HDIM, b_c32, fbuf, HDIM, HDIM);
  rowargmax<<<dim3(BATCH), 256, 0, stream>>>(fbuf, HDIM, k3);

  // ---- main path: GEMM -> kwta (f16 out) x3, final GEMM -> f32 d_out ----
  gemm_f16_wmma<0><<<gemm_grid(BATCH, HID2), 256, 0, stream>>>(
      ctx16, KPAD, wl1, KPAD, b_l1, fbuf, HID2, KPAD);
  kwta_kernel<<<dim3(BATCH), 256, 0, stream>>>(fbuf, k1, h16, HID2); // x1 -> h16

  gemm_f16_wmma<0><<<gemm_grid(BATCH, HID), 256, 0, stream>>>(
      h16, HID2, wl2, HID2, b_l2, fbuf, HID, HID2);
  kwta_kernel<<<dim3(BATCH), 256, 0, stream>>>(fbuf, k2, x2_16, HID);

  gemm_f16_wmma<0><<<gemm_grid(BATCH, HDIM), 256, 0, stream>>>(
      x2_16, HID, wl3, HID, b_l3, fbuf, HDIM, HID);
  kwta_kernel<<<dim3(BATCH), 256, 0, stream>>>(fbuf, k3, x3_16, HDIM);

  gemm_f16_wmma<0><<<gemm_grid(BATCH, HDIM), 256, 0, stream>>>(
      x3_16, HDIM, wl4, HDIM, b_l4, d_out, HDIM, HDIM);
}
